// RegionAttention_54528904790350
// MI455X (gfx1250) — compile-verified, matrix operand order checked by reading, not run on hardware
//
#include <hip/hip_runtime.h>
#include <math.h>

typedef _Float16 half_t;
typedef __attribute__((ext_vector_type(16))) _Float16 v16h;
typedef __attribute__((ext_vector_type(8)))  _Float16 v8h;
typedef __attribute__((ext_vector_type(4)))  _Float16 v4h;
typedef __attribute__((ext_vector_type(8)))  float    v8f;

__device__ __forceinline__ v8f wmma_f16(v16h a, v16h b, v8f c) {
    return __builtin_amdgcn_wmma_f32_16x16x32_f16(false, a, false, b, (short)0, c, false, false);
}

__device__ __forceinline__ v8f v8f_zero() {
    v8f z;
#pragma unroll
    for (int i = 0; i < 8; i++) z[i] = 0.0f;
    return z;
}

// Load one 16x32 f16 WMMA operand tile from LDS (A layout; B is symmetric when
// its rows are the N dimension stored K-major).
// lanes 0-15: row M=lane, K={0..7,16..23}; lanes 16-31: same rows, K={8..15,24..31}
__device__ __forceinline__ v16h load_op(const half_t* tile, int lane, int stride) {
    const half_t* p = tile + (lane & 15) * stride + ((lane >> 4) << 3);
    v8h lo = *(const v8h*)(p);
    v8h hi = *(const v8h*)(p + 16);
    v16h r;
#pragma unroll
    for (int i = 0; i < 8; i++) { r[i] = lo[i]; r[i + 8] = hi[i]; }
    return r;
}

// region-major row index -> original token index
__device__ __forceinline__ int m2l(int m) {
    int rg = m >> 10, n = m & 1023;
    int ib = rg >> 2, jb = rg & 3, ii = n >> 5, jj = n & 31;
    return (ib * 32 + ii) * 128 + jb * 32 + jj;
}

// ---------------------------------------------------------------------------
// Kernel 1: qkv = x @ qkv_w^T + qkv_b (rows gathered region-major, f16 out)
// M=16384, N=1536, K=512. 128x128 tile, 8 waves * (2x4) WMMA tiles.
// Register-prefetch pipelined staging; output staged via LDS for b128 stores.
// ---------------------------------------------------------------------------
__global__ __launch_bounds__(256) void qkv_gemm(const float* __restrict__ x,
                                                const float* __restrict__ w,
                                                const float* __restrict__ bias,
                                                half_t* __restrict__ out) {
    __shared__ half_t As[128 * 40];
    __shared__ half_t Bs[128 * 40];
    __shared__ half_t Cs[128 * 136];
    const int tid = threadIdx.x, lane = tid & 31, wv = tid >> 5;
    const int mbase = blockIdx.x * 128;
    const int nbase = blockIdx.y * 128;
    const int rt0 = (wv >> 1) * 2;
    const int ct0 = (wv & 1) * 4;
    v8f acc[2][4];
#pragma unroll
    for (int i = 0; i < 2; i++)
#pragma unroll
        for (int j = 0; j < 4; j++) acc[i][j] = v8f_zero();

    float4 ar[4], br[4];
#pragma unroll
    for (int i = 0; i < 4; i++) {
        int idx = tid + i * 256;
        int r = idx >> 3, k = (idx & 7) * 4;
        ar[i] = *(const float4*)(x + m2l(mbase + r) * 512 + k);
        br[i] = *(const float4*)(w + (nbase + r) * 512 + k);
    }

    for (int kc = 0; kc < 512; kc += 32) {
        __syncthreads();
#pragma unroll
        for (int i = 0; i < 4; i++) {
            int idx = tid + i * 256;
            int r = idx >> 3, k = (idx & 7) * 4;
            v4h av = {(half_t)ar[i].x, (half_t)ar[i].y, (half_t)ar[i].z, (half_t)ar[i].w};
            v4h bv = {(half_t)br[i].x, (half_t)br[i].y, (half_t)br[i].z, (half_t)br[i].w};
            *(v4h*)(As + r * 40 + k) = av;
            *(v4h*)(Bs + r * 40 + k) = bv;
        }
        __syncthreads();
        if (kc + 32 < 512) {
#pragma unroll
            for (int i = 0; i < 4; i++) {
                int idx = tid + i * 256;
                int r = idx >> 3, k = (idx & 7) * 4;
                ar[i] = *(const float4*)(x + m2l(mbase + r) * 512 + kc + 32 + k);
                br[i] = *(const float4*)(w + (nbase + r) * 512 + kc + 32 + k);
            }
        }
        v16h a0 = load_op(As + (rt0 * 16) * 40, lane, 40);
        v16h a1 = load_op(As + (rt0 * 16 + 16) * 40, lane, 40);
#pragma unroll
        for (int ct = 0; ct < 4; ct++) {
            v16h bm = load_op(Bs + ((ct0 + ct) * 16) * 40, lane, 40);
            acc[0][ct] = wmma_f16(a0, bm, acc[0][ct]);
            acc[1][ct] = wmma_f16(a1, bm, acc[1][ct]);
        }
    }

    // stage C (+bias) into LDS, then coalesced b128 global stores
    const int hi8 = (lane >= 16) ? 8 : 0;
    __syncthreads();
#pragma unroll
    for (int i = 0; i < 2; i++) {
        int rowloc = (rt0 + i) * 16 + hi8;
#pragma unroll
        for (int ct = 0; ct < 4; ct++) {
            int col = (ct0 + ct) * 16 + (lane & 15);
            float bv = bias[nbase + col];
#pragma unroll
            for (int g = 0; g < 8; g++)
                Cs[(rowloc + g) * 136 + col] = (half_t)(acc[i][ct][g] + bv);
        }
    }
    __syncthreads();
    for (int idx = tid; idx < 128 * 16; idx += 256) {
        int r = idx >> 4, ch = idx & 15;
        *(v8h*)(out + (mbase + r) * 1536 + nbase + ch * 8) = *(const v8h*)(Cs + r * 136 + ch * 8);
    }
}

// ---------------------------------------------------------------------------
// Kernel 2: fused region attention with EPEG conv (flash-style over key tiles)
// grid: 16 regions * 8 heads * 8 query tiles. 256 threads = 8 waves.
// Query tile = 128 rows + one 16-row halo tile each side (conv needs i-2..i+2).
// ---------------------------------------------------------------------------
#define QP 72    // Q/K/Ostage LDS pitch (halves)
#define SP 136   // S/P/Vt LDS pitch (halves)

__global__ __launch_bounds__(256) void region_attn(const half_t* __restrict__ qkvh,
                                                   const float* __restrict__ epw,
                                                   const float* __restrict__ epb,
                                                   half_t* __restrict__ ao) {
    extern __shared__ char smemraw[];
    half_t* Qs = (half_t*)smemraw;        // 160 x QP  (also reused as output stage)
    half_t* Ks = Qs + 160 * QP;           // 128 x QP
    half_t* Vt = Ks + 128 * QP;           //  64 x SP (V transposed: [d][j])
    half_t* Ss = Vt + 64 * SP;            // 160 x SP (f16 logits)
    half_t* Ps = Ss + 160 * SP;           // 128 x SP (f16 logits -> probs)
    float* rowscale = (float*)(Ps + 128 * SP);  // [128]
    float* rowsum = rowscale + 128;             // [128]

    const int tid = threadIdx.x, lane = tid & 31, wv = tid >> 5;
    const int rg = blockIdx.x >> 6;
    const int h = (blockIdx.x >> 3) & 7;
    const int qt = blockIdx.x & 7;
    const int qbase = qt * 128;
    const int hi8 = (lane >= 16) ? 8 : 0;

    const half_t* regq = qkvh + rg * 1024 * 1536 + h * 64;

    // ---- stage Q (rows qbase-16 .. qbase+143, pre-scaled by HEAD_DIM^-0.5) ----
    const float scale = 0.125f;
    for (int idx = tid; idx < 160 * 8; idx += 256) {
        int r = idx >> 3, ch = idx & 7;
        int i = qbase - 16 + r;
        v8h qv;
        if (i >= 0 && i < 1024) {
            v8h src = *(const v8h*)(regq + i * 1536 + ch * 8);
#pragma unroll
            for (int e = 0; e < 8; e++) qv[e] = (half_t)((float)src[e] * scale);
        } else {
#pragma unroll
            for (int e = 0; e < 8; e++) qv[e] = (half_t)0.0f;
        }
        *(v8h*)(Qs + r * QP + ch * 8) = qv;
    }

    float cw[5];
#pragma unroll
    for (int t = 0; t < 5; t++) cw[t] = epw[h * 5 + t];
    const float cb = epb[h];

    float m_run = -INFINITY, s_run = 0.0f;
    const int srow = tid >> 1;    // softmax row (0..127)
    const int shalf = tid & 1;    // which 64-col half
    v8f oacc[4];
#pragma unroll
    for (int ct = 0; ct < 4; ct++) oacc[ct] = v8f_zero();

    for (int jt = 0; jt < 8; jt++) {
        const int j0 = jt * 128;
        __syncthreads();
        // ---- stage K tile (b128) and transposed V tile ----
        for (int idx = tid; idx < 128 * 8; idx += 256) {
            int r = idx >> 3, ch = idx & 7;
            const half_t* src = qkvh + (rg * 1024 + j0 + r) * 1536 + 512 + h * 64 + ch * 8;
            *(v8h*)(Ks + r * QP + ch * 8) = *(const v8h*)(src);
        }
        for (int idx = tid; idx < 128 * 8; idx += 256) {
            int r = idx >> 3, ch = idx & 7;
            v8h vv = *(const v8h*)(qkvh + (rg * 1024 + j0 + r) * 1536 + 1024 + h * 64 + ch * 8);
#pragma unroll
            for (int e = 0; e < 8; e++) Vt[(ch * 8 + e) * SP + r] = vv[e];
        }
        __syncthreads();

        // ---- S = Q K^T : wave wv owns a 16-col strip, 10 row tiles (incl. halo) ----
        {
            v16h b0 = load_op(Ks + (wv * 16) * QP + 0, lane, QP);
            v16h b1 = load_op(Ks + (wv * 16) * QP + 32, lane, QP);
            const int col = wv * 16 + (lane & 15);
            for (int rt = 0; rt < 10; rt++) {
                v16h a0 = load_op(Qs + (rt * 16) * QP + 0, lane, QP);
                v16h a1 = load_op(Qs + (rt * 16) * QP + 32, lane, QP);
                v8f s = v8f_zero();
                s = wmma_f16(a0, b0, s);
                s = wmma_f16(a1, b1, s);
                int row0 = rt * 16 + hi8;
#pragma unroll
                for (int g = 0; g < 8; g++) Ss[(row0 + g) * SP + col] = (half_t)s[g];
            }
        }
        __syncthreads();

        // ---- EPEG conv along i + online softmax (vectorized v8h LDS access) ----
        {
            const int e = srow + 16;
            const int jb = shalf * 64;
            float lmax = -INFINITY;
#pragma unroll
            for (int c8 = 0; c8 < 8; c8++) {
                int j = jb + c8 * 8;
                const half_t* base = Ss + e * SP + j;
                v8h rm2 = *(const v8h*)(base - 2 * SP);
                v8h rm1 = *(const v8h*)(base - SP);
                v8h r0  = *(const v8h*)(base);
                v8h rp1 = *(const v8h*)(base + SP);
                v8h rp2 = *(const v8h*)(base + 2 * SP);
                v8h lout;
#pragma unroll
                for (int k = 0; k < 8; k++) {
                    float l = (float)r0[k] + cb
                            + cw[0] * (float)rm2[k] + cw[1] * (float)rm1[k]
                            + cw[2] * (float)r0[k]  + cw[3] * (float)rp1[k]
                            + cw[4] * (float)rp2[k];
                    lmax = fmaxf(lmax, l);
                    lout[k] = (half_t)l;
                }
                *(v8h*)(Ps + srow * SP + j) = lout;
            }
            lmax = fmaxf(lmax, __shfl_xor(lmax, 1, 32));
            float m_new = fmaxf(m_run, lmax);
            float sc = __expf(m_run - m_new);
            float ssum = 0.0f;
#pragma unroll
            for (int c8 = 0; c8 < 8; c8++) {
                int j = jb + c8 * 8;
                v8h lv = *(const v8h*)(Ps + srow * SP + j);
                v8h pv;
#pragma unroll
                for (int k = 0; k < 8; k++) {
                    float p = __expf((float)lv[k] - m_new);
                    ssum += p;
                    pv[k] = (half_t)p;
                }
                *(v8h*)(Ps + srow * SP + j) = pv;
            }
            ssum += __shfl_xor(ssum, 1, 32);
            s_run = s_run * sc + ssum;
            m_run = m_new;
            if (shalf == 0) rowscale[srow] = sc;
        }
        __syncthreads();

        // ---- O = rescale(O) + P V : wave wv owns a 16-row tile, 4 d-tiles ----
        {
#pragma unroll
            for (int g = 0; g < 8; g++) {
                float f = rowscale[wv * 16 + g + hi8];
#pragma unroll
                for (int ct = 0; ct < 4; ct++) oacc[ct][g] *= f;
            }
            for (int ks = 0; ks < 4; ks++) {
                v16h a = load_op(Ps + (wv * 16) * SP + ks * 32, lane, SP);
#pragma unroll
                for (int ct = 0; ct < 4; ct++) {
                    v16h bm = load_op(Vt + (ct * 16) * SP + ks * 32, lane, SP);
                    oacc[ct] = wmma_f16(a, bm, oacc[ct]);
                }
            }
        }
    }

    if (shalf == 0) rowsum[srow] = s_run;
    __syncthreads();   // also: Qs free from here on -> reuse as output stage

    // ---- normalize into LDS stage, then coalesced b128 stores ----
#pragma unroll
    for (int ct = 0; ct < 4; ct++) {
#pragma unroll
        for (int g = 0; g < 8; g++) {
            int row = wv * 16 + g + hi8;
            float inv = 1.0f / rowsum[row];
            Qs[row * QP + ct * 16 + (lane & 15)] = (half_t)(oacc[ct][g] * inv);
        }
    }
    __syncthreads();
    for (int idx = tid; idx < 128 * 8; idx += 256) {
        int r = idx >> 3, ch = idx & 7;
        *(v8h*)(ao + (rg * 1024 + qbase + r) * 512 + h * 64 + ch * 8) =
            *(const v8h*)(Qs + r * QP + ch * 8);
    }
}

// ---------------------------------------------------------------------------
// Kernel 3: out = ao @ proj_w^T + proj_b, scattered back to token order (f32)
// M=16384, N=512, K=512. Output staged through LDS in two 64-row halves.
// ---------------------------------------------------------------------------
__global__ __launch_bounds__(256) void proj_gemm(const half_t* __restrict__ ao,
                                                 const float* __restrict__ w,
                                                 const float* __restrict__ bias,
                                                 float* __restrict__ out) {
    __shared__ half_t As[128 * 40];
    __shared__ half_t Bs[128 * 40];
    __shared__ float Cs32[64 * 132];
    const int tid = threadIdx.x, lane = tid & 31, wv = tid >> 5;
    const int mbase = blockIdx.x * 128;
    const int nbase = blockIdx.y * 128;
    const int rt0 = (wv >> 1) * 2;
    const int ct0 = (wv & 1) * 4;
    v8f acc[2][4];
#pragma unroll
    for (int i = 0; i < 2; i++)
#pragma unroll
        for (int j = 0; j < 4; j++) acc[i][j] = v8f_zero();

    v8h areg[2];
    float4 brg[4];
#pragma unroll
    for (int i = 0; i < 2; i++) {
        int idx = tid + i * 256;
        int r = idx >> 2, k = (idx & 3) * 8;
        areg[i] = *(const v8h*)(ao + (mbase + r) * 512 + k);
    }
#pragma unroll
    for (int i = 0; i < 4; i++) {
        int idx = tid + i * 256;
        int r = idx >> 3, k = (idx & 7) * 4;
        brg[i] = *(const float4*)(w + (nbase + r) * 512 + k);
    }

    for (int kc = 0; kc < 512; kc += 32) {
        __syncthreads();
#pragma unroll
        for (int i = 0; i < 2; i++) {
            int idx = tid + i * 256;
            int r = idx >> 2, k = (idx & 3) * 8;
            *(v8h*)(As + r * 40 + k) = areg[i];
        }
#pragma unroll
        for (int i = 0; i < 4; i++) {
            int idx = tid + i * 256;
            int r = idx >> 3, k = (idx & 7) * 4;
            v4h bv = {(half_t)brg[i].x, (half_t)brg[i].y, (half_t)brg[i].z, (half_t)brg[i].w};
            *(v4h*)(Bs + r * 40 + k) = bv;
        }
        __syncthreads();
        if (kc + 32 < 512) {
#pragma unroll
            for (int i = 0; i < 2; i++) {
                int idx = tid + i * 256;
                int r = idx >> 2, k = (idx & 3) * 8;
                areg[i] = *(const v8h*)(ao + (mbase + r) * 512 + kc + 32 + k);
            }
#pragma unroll
            for (int i = 0; i < 4; i++) {
                int idx = tid + i * 256;
                int r = idx >> 3, k = (idx & 7) * 4;
                brg[i] = *(const float4*)(w + (nbase + r) * 512 + kc + 32 + k);
            }
        }
        v16h a0 = load_op(As + (rt0 * 16) * 40, lane, 40);
        v16h a1 = load_op(As + (rt0 * 16 + 16) * 40, lane, 40);
#pragma unroll
        for (int ct = 0; ct < 4; ct++) {
            v16h bm = load_op(Bs + ((ct0 + ct) * 16) * 40, lane, 40);
            acc[0][ct] = wmma_f16(a0, bm, acc[0][ct]);
            acc[1][ct] = wmma_f16(a1, bm, acc[1][ct]);
        }
    }

    const int hi8 = (lane >= 16) ? 8 : 0;
    __syncthreads();
#pragma unroll
    for (int hb = 0; hb < 2; hb++) {
        int rowbase = hb * 64;
        if (rt0 * 16 >= rowbase && rt0 * 16 < rowbase + 64) {
#pragma unroll
            for (int i = 0; i < 2; i++) {
                int rowloc = (rt0 + i) * 16 + hi8 - rowbase;
#pragma unroll
                for (int ct = 0; ct < 4; ct++) {
                    int col = (ct0 + ct) * 16 + (lane & 15);
                    float bv = bias[nbase + col];
#pragma unroll
                    for (int g = 0; g < 8; g++)
                        Cs32[(rowloc + g) * 132 + col] = acc[i][ct][g] + bv;
                }
            }
        }
        __syncthreads();
        for (int idx = tid; idx < 64 * 32; idx += 256) {
            int r = idx >> 5, ch = idx & 31;
            int l = m2l(mbase + rowbase + r);
            *(float4*)(out + l * 512 + nbase + ch * 4) = *(const float4*)(Cs32 + r * 132 + ch * 4);
        }
        __syncthreads();
    }
}

extern "C" void kernel_launch(void* const* d_in, const int* in_sizes, int n_in,
                              void* d_out, int out_size, void* d_ws, size_t ws_size,
                              hipStream_t stream) {
    (void)in_sizes; (void)n_in; (void)out_size; (void)ws_size;
    const float* x      = (const float*)d_in[0];
    const float* qkv_w  = (const float*)d_in[1];
    const float* qkv_b  = (const float*)d_in[2];
    const float* proj_w = (const float*)d_in[3];
    const float* proj_b = (const float*)d_in[4];
    const float* epeg_w = (const float*)d_in[5];
    const float* epeg_b = (const float*)d_in[6];
    float* out = (float*)d_out;

    half_t* qkvh = (half_t*)d_ws;                      // [16384][1536] f16 (region-major rows)
    half_t* ao = qkvh + (size_t)16384 * 1536;          // [16384][512]  f16

    qkv_gemm<<<dim3(128, 12), 256, 0, stream>>>(x, qkv_w, qkv_b, qkvh);

    size_t shbytes = (size_t)(160 * QP + 128 * QP + 64 * SP + 160 * SP + 128 * SP) * sizeof(half_t)
                   + 256 * sizeof(float);              // = 138240 bytes
    region_attn<<<dim3(16 * 8 * 8), 256, shbytes, stream>>>(qkvh, epeg_w, epeg_b, ao);

    proj_gemm<<<dim3(128, 4), 256, 0, stream>>>(ao, proj_w, proj_b, out);
}
